// Attention_61366492725852
// MI455X (gfx1250) — matrix operand order repaired and checked by reading, not compiled, not run on hardware
//
#include <hip/hip_runtime.h>
#include <math.h>

#define DIM   256
#define KLEN  2048
#define BATCH 256

typedef __attribute__((ext_vector_type(16))) __bf16 v16bf;
typedef __attribute__((ext_vector_type(8)))  __bf16 v8bf;
typedef __attribute__((ext_vector_type(2)))  __bf16 v2bf;
typedef __attribute__((ext_vector_type(8)))  float  v8f;

__device__ __forceinline__ float fast_tanh(float x) {
#if __has_builtin(__builtin_amdgcn_tanhf)
    return __builtin_amdgcn_tanhf(x);
#elif __has_builtin(__builtin_amdgcn_tanh_f32)
    return __builtin_amdgcn_tanh_f32(x);
#else
    return tanhf(x);
#endif
}

// Convert 16 f32 -> v16bf, preferring packed hardware converts.
__device__ __forceinline__ v16bf cvt16_bf16(const float* xs) {
    v16bf t;
#if __has_builtin(__builtin_amdgcn_cvt_pk_bf16_f32)
#pragma unroll
    for (int j = 0; j < 8; ++j) {
        v2bf p = __builtin_amdgcn_cvt_pk_bf16_f32(xs[2 * j], xs[2 * j + 1]);
        t[2 * j]     = p[0];
        t[2 * j + 1] = p[1];
    }
#else
#pragma unroll
    for (int j = 0; j < 16; ++j) t[j] = (__bf16)xs[j];
#endif
    return t;
}

// ---------------------------------------------------------------------------
// Kernel 1: qb[b][d] = sum_e query[b][e] * Wq[d][e] + bq[d] + bref[d]
// ---------------------------------------------------------------------------
__global__ void qprep_kernel(const float* __restrict__ query,
                             const float* __restrict__ Wq,
                             const float* __restrict__ bq,
                             const float* __restrict__ bref,
                             float* __restrict__ qb) {
    __shared__ float sq[DIM];
    const int b = blockIdx.x;
    const int d = threadIdx.x;
    sq[d] = query[b * DIM + d];
    __syncthreads();

    const float4* w4 = (const float4*)(Wq + (size_t)d * DIM);
    const float4* q4 = (const float4*)sq;
    float acc = 0.f;
#pragma unroll 8
    for (int e = 0; e < DIM / 4; ++e) {
        float4 a = w4[e], s = q4[e];
        acc = fmaf(a.x, s.x, acc);
        acc = fmaf(a.y, s.y, acc);
        acc = fmaf(a.z, s.z, acc);
        acc = fmaf(a.w, s.w, acc);
    }
    qb[b * DIM + d] = acc + bq[d] + bref[d];
}

// ---------------------------------------------------------------------------
// Kernel 2: fused GEMM (Wref @ enc) + tanh/v reduction + logit clip.
//   grid = (KLEN/32, BATCH/16); block = 256 threads (8 waves).
//   Each wave carries TWO k-columns at once (two resident B-fragment sets),
//   so every Wref A-fragment fetched from LDS feeds two WMMAs.
// ---------------------------------------------------------------------------
#define WREF_STRIDE 264   // bf16 elems/row: 528B, 16B aligned, bank-skewed
#define QB_STRIDE   260   // f32 elems/row: 1040B, 16B aligned

__global__ __launch_bounds__(256, 1)
void attn_kernel(const float* __restrict__ enc,
                 const float* __restrict__ Wref,
                 const float* __restrict__ v,
                 const float* __restrict__ qb,
                 float* __restrict__ out) {
    __shared__ __bf16 sW[DIM * WREF_STRIDE];   // ~132 KB: Wref in bf16
    __shared__ float  sQ[16 * QB_STRIDE];      // ~16.6 KB: qb tile for b-block
    __shared__ float  sV[DIM];                 // 1 KB

    const int tid = threadIdx.x;
    const int b0  = blockIdx.y * 16;

    for (int idx = tid; idx < DIM * DIM; idx += 256) {
        const int r = idx >> 8, c = idx & 255;
        sW[r * WREF_STRIDE + c] = (__bf16)Wref[idx];
    }
    for (int idx = tid; idx < 16 * DIM; idx += 256) {
        const int r = idx >> 8, c = idx & 255;
        sQ[r * QB_STRIDE + c] = qb[(size_t)(b0 + r) * DIM + c];
    }
    sV[tid] = v[tid];
    __syncthreads();

    const int wave = tid >> 5;
    const int lane = tid & 31;
    const int bcol = lane & 15;   // N index (column within b-block)
    const int hi   = lane >> 4;   // half-wave selector
    const int k0   = blockIdx.x * 32;

    for (int it = 0; it < 2; ++it) {
        const int kA = k0 + wave + 16 * it;   // first k-column
        const int kB = kA + 8;                // second k-column

        // --- Build both B-fragment sets (f32 -> bf16).
        // B layout (32x16 bf16): lane = N; lanes 0-15 hold K=0..15, lanes
        // 16-31 hold K=16..31 -> contiguous 16 e-values per lane per chunk.
        const float* encA =
            enc + ((size_t)kA * BATCH + (b0 + bcol)) * DIM + hi * 16;
        const float* encB =
            enc + ((size_t)kB * BATCH + (b0 + bcol)) * DIM + hi * 16;
        v16bf bfragA[8], bfragB[8];
#pragma unroll
        for (int c = 0; c < 8; ++c) {
            const float4* pa = (const float4*)(encA + c * 32);
            float4 a0 = pa[0], a1 = pa[1], a2 = pa[2], a3 = pa[3];
            const float xa[16] = {a0.x, a0.y, a0.z, a0.w, a1.x, a1.y, a1.z, a1.w,
                                  a2.x, a2.y, a2.z, a2.w, a3.x, a3.y, a3.z, a3.w};
            bfragA[c] = cvt16_bf16(xa);
            const float4* pb = (const float4*)(encB + c * 32);
            float4 b0v = pb[0], b1v = pb[1], b2v = pb[2], b3v = pb[3];
            const float xb[16] = {b0v.x, b0v.y, b0v.z, b0v.w, b1v.x, b1v.y, b1v.z, b1v.w,
                                  b2v.x, b2v.y, b2v.z, b2v.w, b3v.x, b3v.y, b3v.z, b3v.w};
            bfragB[c] = cvt16_bf16(xb);
        }

        // Prefetch the next pass's tiles (global_prefetch_b8).
        if (it == 0) {
            __builtin_prefetch(encA + (size_t)16 * BATCH * DIM, 0, 1);
            __builtin_prefetch(encB + (size_t)16 * BATCH * DIM, 0, 1);
        }

        // --- 16 m-tiles; qb folded into the WMMA C operand.
        float uA = 0.f, uB = 0.f;
        for (int m = 0; m < 16; ++m) {
            // This lane's C/D rows are d = m*16 + hi*8 + j  (j = 0..7).
            const float* qp = &sQ[bcol * QB_STRIDE + m * 16 + hi * 8];
            float4 q0 = *(const float4*)qp, q1 = *(const float4*)(qp + 4);
            v8f accA = {q0.x, q0.y, q0.z, q0.w, q1.x, q1.y, q1.z, q1.w};
            v8f accB = accA;

            // A layout (16x32 bf16): lane%16 = M; contiguous K = hi*16 .. hi*16+15.
            const __bf16* arow = &sW[(m * 16 + bcol) * WREF_STRIDE + hi * 16];
#pragma unroll
            for (int c = 0; c < 8; ++c) {
                v8bf a0 = *(const v8bf*)(arow + c * 32);
                v8bf a1 = *(const v8bf*)(arow + c * 32 + 8);
                v16bf af = __builtin_shufflevector(
                    a0, a1, 0, 1, 2, 3, 4, 5, 6, 7, 8, 9, 10, 11, 12, 13, 14, 15);
                accA = __builtin_amdgcn_wmma_f32_16x16x32_bf16(
                    false, af, false, bfragA[c], (short)0, accA, false, false);
                accB = __builtin_amdgcn_wmma_f32_16x16x32_bf16(
                    false, af, false, bfragB[c], (short)0, accB, false, false);
            }

            const float* vp = &sV[m * 16 + hi * 8];
            float4 w0 = *(const float4*)vp, w1 = *(const float4*)(vp + 4);
            const float ws[8] = {w0.x, w0.y, w0.z, w0.w, w1.x, w1.y, w1.z, w1.w};
#pragma unroll
            for (int j = 0; j < 8; ++j) {
                uA = fmaf(ws[j], fast_tanh(accA[j]), uA);
                uB = fmaf(ws[j], fast_tanh(accB[j]), uB);
            }
        }

        // Combine the two half-wave row groups (wave32 xor-shuffle by 16).
        uA += __shfl_xor(uA, 16, 32);
        uB += __shfl_xor(uB, 16, 32);
        if (lane < 16) {
            out[(size_t)(b0 + bcol) * KLEN + kA] = 10.0f * fast_tanh(uA);
            out[(size_t)(b0 + bcol) * KLEN + kB] = 10.0f * fast_tanh(uB);
        }
    }
}

// ---------------------------------------------------------------------------
extern "C" void kernel_launch(void* const* d_in, const int* in_sizes, int n_in,
                              void* d_out, int out_size, void* d_ws, size_t ws_size,
                              hipStream_t stream) {
    const float* enc   = (const float*)d_in[0];  // (K, B, DIM)
    const float* query = (const float*)d_in[1];  // (B, DIM)
    const float* Wq    = (const float*)d_in[2];  // (DIM, DIM)
    const float* bq    = (const float*)d_in[3];  // (DIM,)
    const float* Wref  = (const float*)d_in[4];  // (DIM, DIM)
    const float* bref  = (const float*)d_in[5];  // (DIM,)
    const float* v     = (const float*)d_in[6];  // (DIM,)
    float* out = (float*)d_out;                  // (B, K) f32
    float* qb  = (float*)d_ws;                   // 256 KB scratch: qb[b][d]

    qprep_kernel<<<BATCH, DIM, 0, stream>>>(query, Wq, bq, bref, qb);
    attn_kernel<<<dim3(KLEN / 32, BATCH / 16), 256, 0, stream>>>(
        enc, Wref, v, qb, out);
}
